// SimpleSpConvNet_51187420234529
// MI455X (gfx1250) — matrix operand order repaired and checked by reading, not compiled, not run on hardware
//
#include <hip/hip_runtime.h>

typedef __attribute__((ext_vector_type(16))) __bf16 v16bf;
typedef __attribute__((ext_vector_type(8)))  float  v8f;

#define DIM   65
#define D3    (65*65*65)
#define S     32
#define S3    (32*32*32)
#define PD    34
#define PD3   (34*34*34)
#define CO    100
#define EPS   1e-5f

// ---------- helpers ----------
__device__ __forceinline__ unsigned short f2bf(float f) {
    unsigned int u = __float_as_uint(f);
    u += 0x7FFFu + ((u >> 16) & 1u);          // round-to-nearest-even
    return (unsigned short)(u >> 16);
}

// ---------- zero fill (h2pad halo init) ----------
__global__ void fill_zero_u4(uint4* __restrict__ p, int n4) {
    int i = blockIdx.x * blockDim.x + threadIdx.x;
    if (i < n4) p[i] = make_uint4(0u, 0u, 0u, 0u);
}

// ---------- weight swizzle into WMMA 16-bit A layout ----------
// A tile element order per lane (16 bf16 = 8 VGPR dwords):
//   K = 16*half + 2j + p
// A2: [27 taps][2 mtiles][32 lanes][16]   (W2: 32x32x27)
// Ai: [27 taps][7 mtiles][32 lanes][16]   (Winv: 100x32x27, rows >=100 zero)
__global__ void prep_weights(const float* __restrict__ W2,
                             const float* __restrict__ Winv,
                             unsigned short* __restrict__ A2,
                             unsigned short* __restrict__ Ai) {
    int tid = blockIdx.x * blockDim.x + threadIdx.x;
    const int n2 = 27 * 2 * 32;
    const int ni = 27 * 7 * 32;
    if (tid < n2) {
        int lane = tid & 31, mt = (tid >> 5) & 1, t = tid >> 6;
        int half = lane >> 4, m = lane & 15;
        int co = mt * 16 + m;
        unsigned short* dst = A2 + (size_t)tid * 16;
        #pragma unroll
        for (int e = 0; e < 16; ++e) {
            int j = e >> 1, p = e & 1;
            int k = 16 * half + 2 * j + p;
            dst[e] = f2bf(W2[(co * 32 + k) * 27 + t]);
        }
    } else if (tid < n2 + ni) {
        int id = tid - n2;
        int lane = id & 31, mt = (id >> 5) % 7, t = (id >> 5) / 7;
        int half = lane >> 4, m = lane & 15;
        int co = mt * 16 + m;
        unsigned short* dst = Ai + (size_t)id * 16;
        #pragma unroll
        for (int e = 0; e < 16; ++e) {
            int j = e >> 1, p = e & 1;
            int k = 16 * half + 2 * j + p;
            dst[e] = (co < CO) ? f2bf(Winv[(co * 32 + k) * 27 + t]) : (unsigned short)0;
        }
    }
}

// ---------- conv1 (Cin=1, k=3, s=2) + BN + ReLU + mask, writes padded bf16 ----------
__global__ void conv1_bn_relu(const float* __restrict__ x, const int* __restrict__ mask0,
                              const float* __restrict__ W1, const float* __restrict__ b1,
                              const float* __restrict__ g1, const float* __restrict__ bt1,
                              const float* __restrict__ rm1, const float* __restrict__ rv1,
                              unsigned short* __restrict__ h1pad, float* __restrict__ m1) {
    int tid = blockIdx.x * blockDim.x + threadIdx.x;
    if (tid >= 2 * PD3) return;
    int b = tid / PD3, r = tid % PD3;
    int zp = r / (PD * PD), yp = (r / PD) % PD, xp = r % PD;
    unsigned short* dst = h1pad + (size_t)tid * 32;
    if (zp == 0 || zp == PD - 1 || yp == 0 || yp == PD - 1 || xp == 0 || xp == PD - 1) {
        #pragma unroll
        for (int c = 0; c < 32; ++c) dst[c] = 0;
        return;
    }
    int z = zp - 1, y = yp - 1, xx = xp - 1;
    float xv[27];
    int cnt = 0;
    #pragma unroll
    for (int kz = 0; kz < 3; ++kz)
        #pragma unroll
        for (int ky = 0; ky < 3; ++ky)
            #pragma unroll
            for (int kx = 0; kx < 3; ++kx) {
                int idx = b * D3 + ((2 * z + kz) * DIM + (2 * y + ky)) * DIM + (2 * xx + kx);
                xv[(kz * 3 + ky) * 3 + kx] = x[idx];
                cnt += mask0[idx];
            }
    float mval = (cnt > 0) ? 1.f : 0.f;
    m1[((b * S + z) * S + y) * S + xx] = mval;
    for (int c = 0; c < 32; ++c) {
        float acc = b1[c];
        #pragma unroll
        for (int t = 0; t < 27; ++t) acc += xv[t] * W1[c * 27 + t];
        float inv = g1[c] * rsqrtf(rv1[c] + EPS);
        float h = (acc - rm1[c]) * inv + bt1[c];
        h = fmaxf(h, 0.f) * mval;
        dst[c] = f2bf(h);
    }
}

// ---------- conv2 (SubM 32->32, k=3) via WMMA bf16, + BN + ReLU + mask ----------
// One wave computes a 32(Cout) x 16(voxel) tile: 27 taps x 2 WMMA.
__global__ void __launch_bounds__(256)
conv2_wmma(const unsigned short* __restrict__ h1pad,
           const unsigned short* __restrict__ A2,
           const float* __restrict__ m1,
           const float* __restrict__ b2, const float* __restrict__ g2,
           const float* __restrict__ bt2, const float* __restrict__ rm2,
           const float* __restrict__ rv2,
           unsigned short* __restrict__ h2pad) {
    int w    = blockIdx.x * 8 + (threadIdx.x >> 5);    // 4096 waves exactly
    int lane = threadIdx.x & 31;
    int half = lane >> 4, col = lane & 15;
    int xt = w & 1, y = (w >> 1) & 31, z = (w >> 6) & 31, b = w >> 11;
    int x = xt * 16 + col;

    v8f c0 = {}, c1 = {};
    const unsigned short* bbase = h1pad + (size_t)b * PD3 * 32;
    #pragma unroll
    for (int kz = 0; kz < 3; ++kz)
        #pragma unroll
        for (int ky = 0; ky < 3; ++ky) {
            const unsigned short* rowp = bbase + (size_t)((z + kz) * PD + (y + ky)) * PD * 32;
            #pragma unroll
            for (int kx = 0; kx < 3; ++kx) {
                int t = (kz * 3 + ky) * 3 + kx;
                v16bf bf = *(const v16bf*)(rowp + (x + kx) * 32 + 16 * half);
                v16bf a0 = *(const v16bf*)(A2 + (size_t)((t * 2 + 0) * 32 + lane) * 16);
                v16bf a1 = *(const v16bf*)(A2 + (size_t)((t * 2 + 1) * 32 + lane) * 16);
                c0 = __builtin_amdgcn_wmma_f32_16x16x32_bf16(false, a0, false, bf, (short)0, c0, false, false);
                c1 = __builtin_amdgcn_wmma_f32_16x16x32_bf16(false, a1, false, bf, (short)0, c1, false, false);
            }
        }
    float mval = m1[((b * S + z) * S + y) * S + x];
    unsigned short* dvox = h2pad + ((((size_t)b * PD + (z + 1)) * PD + (y + 1)) * PD + (x + 1)) * 32;
    #pragma unroll
    for (int mt = 0; mt < 2; ++mt) {
        v8f acc = mt ? c1 : c0;
        #pragma unroll
        for (int r2 = 0; r2 < 8; ++r2) {
            int co = mt * 16 + 8 * half + r2;
            float inv = g2[co] * rsqrtf(rv2[co] + EPS);
            float h = ((acc[r2] + b2[co]) - rm2[co]) * inv + bt2[co];
            h = fmaxf(h, 0.f) * mval;
            dvox[co] = f2bf(h);
        }
    }
}

// ---------- inverse conv (32->100, transposed s=2) via WMMA bf16 gather ----------
// One wave computes a 112(Cout) x 16(xo) tile for fixed (b, zo, yo).
__global__ void __launch_bounds__(256)
conv3_wmma(const unsigned short* __restrict__ h2pad,
           const unsigned short* __restrict__ Ai,
           const int* __restrict__ mask0,
           const float* __restrict__ binv,
           float* __restrict__ out) {
    const int NW = 2 * 65 * 65 * 5;
    int w = blockIdx.x * 8 + (threadIdx.x >> 5);
    if (w >= NW) w = 0;                       // uniform clamp: keep EXEC all-1 for WMMA
    int lane = threadIdx.x & 31;
    int half = lane >> 4, col = lane & 15;
    int xt = w % 5, yo = (w / 5) % 65, zo = (w / 325) % 65, b = w / 21125;
    int xo = xt * 16 + col;

    v8f acc[7];
    v8f vzero = {};
    #pragma unroll
    for (int i = 0; i < 7; ++i) acc[i] = vzero;

    const unsigned short* bbase = h2pad + (size_t)b * PD3 * 32;
    for (int kz = 0; kz < 3; ++kz) {
        int z2 = zo + kz - 2;
        if (z2 & 1) continue;
        int zi = z2 >> 1;
        if (zi < 0 || zi >= 32) continue;
        for (int ky = 0; ky < 3; ++ky) {
            int y2 = yo + ky - 2;
            if (y2 & 1) continue;
            int yi = y2 >> 1;
            if (yi < 0 || yi >= 32) continue;
            const unsigned short* rowp = bbase + (size_t)((zi + 1) * PD + (yi + 1)) * PD * 32;
            #pragma unroll
            for (int kx = 0; kx < 3; ++kx) {
                int x2 = xo + kx - 2;
                bool valid = ((x2 & 1) == 0) && (x2 >= 0) && (x2 < 64) && (xo < 65);
                v16bf bf = {};
                if (valid) bf = *(const v16bf*)(rowp + ((x2 >> 1) + 1) * 32 + 16 * half);
                int t = (kz * 3 + ky) * 3 + kx;
                const unsigned short* abase = Ai + (size_t)(t * 7) * 32 * 16;
                #pragma unroll
                for (int mt = 0; mt < 7; ++mt) {
                    v16bf a = *(const v16bf*)(abase + (size_t)(mt * 32 + lane) * 16);
                    acc[mt] = __builtin_amdgcn_wmma_f32_16x16x32_bf16(false, a, false, bf, (short)0, acc[mt], false, false);
                }
            }
        }
    }
    if (xo < 65) {
        float mv = mask0[b * D3 + (zo * DIM + yo) * DIM + xo] ? 1.f : 0.f;
        #pragma unroll
        for (int mt = 0; mt < 7; ++mt)
            #pragma unroll
            for (int r2 = 0; r2 < 8; ++r2) {
                int co = mt * 16 + 8 * half + r2;
                if (co < CO) {
                    float v = (acc[mt][r2] + binv[co]) * mv;
                    out[(((size_t)(b * 100 + co) * 65 + zo) * 65 + yo) * 65 + xo] = v;
                }
            }
    }
}

// ---------- host launch ----------
extern "C" void kernel_launch(void* const* d_in, const int* in_sizes, int n_in,
                              void* d_out, int out_size, void* d_ws, size_t ws_size,
                              hipStream_t stream) {
    const float* x    = (const float*)d_in[0];
    const int*   mask0= (const int*)  d_in[1];
    const float* W1   = (const float*)d_in[2];
    const float* b1   = (const float*)d_in[3];
    const float* g1   = (const float*)d_in[4];
    const float* bt1  = (const float*)d_in[5];
    const float* rm1  = (const float*)d_in[6];
    const float* rv1  = (const float*)d_in[7];
    const float* W2   = (const float*)d_in[8];
    const float* b2   = (const float*)d_in[9];
    const float* g2   = (const float*)d_in[10];
    const float* bt2  = (const float*)d_in[11];
    const float* rm2  = (const float*)d_in[12];
    const float* rv2  = (const float*)d_in[13];
    const float* Winv = (const float*)d_in[14];
    const float* binv = (const float*)d_in[15];
    float* out = (float*)d_out;

    // workspace carve-up (256B aligned)
    char* ws = (char*)d_ws;
    size_t off = 0;
    auto take = [&](size_t bytes) {
        size_t o = off;
        off = (off + bytes + 255) & ~(size_t)255;
        return o;
    };
    const size_t hbytes = (size_t)2 * PD3 * 32 * 2;      // padded bf16 volume
    unsigned short* h1pad = (unsigned short*)(ws + take(hbytes));
    unsigned short* h2pad = (unsigned short*)(ws + take(hbytes));
    float*          m1    = (float*)         (ws + take((size_t)2 * S3 * 4));
    unsigned short* A2    = (unsigned short*)(ws + take((size_t)27 * 2 * 32 * 16 * 2));
    unsigned short* Ai    = (unsigned short*)(ws + take((size_t)27 * 7 * 32 * 16 * 2));
    (void)ws_size; (void)n_in; (void)in_sizes; (void)out_size;

    // 1) zero h2pad (halo regions never written by conv2)
    int n4 = (int)(hbytes / 16);
    fill_zero_u4<<<(n4 + 255) / 256, 256, 0, stream>>>((uint4*)h2pad, n4);

    // 2) swizzle weights into WMMA A layout (bf16)
    int nprep = 27 * 2 * 32 + 27 * 7 * 32;
    prep_weights<<<(nprep + 255) / 256, 256, 0, stream>>>(W2, Winv, A2, Ai);

    // 3) conv1 + BN + ReLU + mask -> h1pad (bf16, padded), m1
    int n1 = 2 * PD3;
    conv1_bn_relu<<<(n1 + 255) / 256, 256, 0, stream>>>(x, mask0, W1, b1, g1, bt1, rm1, rv1,
                                                        h1pad, m1);

    // 4) conv2 WMMA: 4096 waves, 8 per block
    conv2_wmma<<<512, 256, 0, stream>>>(h1pad, A2, m1, b2, g2, bt2, rm2, rv2, h2pad);

    // 5) inverse conv WMMA: 42250 waves
    int nw3 = 2 * 65 * 65 * 5;
    conv3_wmma<<<(nw3 + 7) / 8, 256, 0, stream>>>(h2pad, Ai, mask0, binv, out);
}
